// MovieNet_25331717111819
// MI455X (gfx1250) — compile-verified, matrix operand order checked by reading, not run on hardware
//
#include <hip/hip_runtime.h>
#include <hip/hip_bf16.h>
#include <math.h>

// ---------------------------------------------------------------------------
// CDNA5 (gfx1250) implementation of the MovieNet-style multimodal LSTM model.
// Matrix work (projections + recurrent h@Whh) uses v_wmma_f32_16x16x32_bf16.
// Projection A-tiles are staged to LDS with GLOBAL_LOAD_ASYNC_TO_LDS_B128
// (ASYNCcnt); weights are pre-transposed to bf16 so B-fragments are
// contiguous 32-byte loads; LSTM hidden state lives in LDS (bf16).
// ---------------------------------------------------------------------------

typedef __attribute__((ext_vector_type(16))) __bf16 v16bf;
typedef __attribute__((ext_vector_type(8)))  float  v8f;

#define Bsz   16
#define Tlen  1024
#define Hdim  256
#define G4H   1024   // 4*H
#define KMAX  768    // largest projection K (text)

__device__ __forceinline__ v8f wmma_bf16(v16bf a, v16bf b, v8f c) {
  // (neg_a, A, neg_b, B, c_mod, C, reuse_a, reuse_b)
  return __builtin_amdgcn_wmma_f32_16x16x32_bf16(false, a, false, b,
                                                 (short)0, c, false, false);
}

// Build a 16-element bf16 fragment from 16 consecutive fp32 values.
__device__ __forceinline__ v16bf frag_from_f32(const float* p) {
  v16bf v;
#pragma unroll
  for (int i = 0; i < 16; ++i) v[i] = (__bf16)p[i];
  return v;
}

__device__ __forceinline__ float sigm(float x) { return 1.0f / (1.0f + __expf(-x)); }

// Async copy of 16 bytes global -> LDS (tracked by ASYNCcnt).
__device__ __forceinline__ void async_copy_b128(const float* gptr, void* lds_ptr) {
  // Low 32 bits of a generic shared-memory pointer == LDS byte address.
  unsigned lds_addr = (unsigned)(unsigned long long)lds_ptr;
  asm volatile("global_load_async_to_lds_b128 %0, %1, off"
               :: "v"(lds_addr), "v"(gptr) : "memory");
}
__device__ __forceinline__ void wait_asynccnt0() {
  asm volatile("s_wait_asynccnt 0" ::: "memory");
}

// ---------------------------------------------------------------------------
// Workspace layout (bytes, all offsets 256B aligned)
// ---------------------------------------------------------------------------
#define OFF_TR   0u          // tr  (16384 x 256) f32   16 MB
#define OFF_FR   16777216u   // fr
#define OFF_AR   33554432u   // ar
#define OFF_SC   50331648u   // scores [6][16][1024] f32
#define OFF_EIN  50724864u   // enc_input (B,T,3) f32
#define OFF_ATT  50921472u   // attn      (B,T,3) f32
#define OFF_EO   51118080u   // enc_out   (B,T,3) f32
#define OFF_CTX  51314688u   // context   (B,T,3) f32
#define OFF_WTT  51511296u   // text_W^T  bf16 (256 x 768)
#define OFF_WFT  51904512u   // face_W^T  bf16 (256 x 512)
#define OFF_WAT  52166656u   // audio_W^T bf16 (256 x 256)
#define OFF_WHE  52297728u   // enc_Whh^T bf16 (3 x 1024 x 256)
#define OFF_WHD  53870592u   // dec_Whh^T bf16 (1024 x 256)

// ---------------------------------------------------------------------------
// Transpose fp32 (K x N) -> bf16 (N x K)   dst[n*K+k] = src[k*N+n]
// ---------------------------------------------------------------------------
__global__ void k_transpose_bf16(const float* __restrict__ src,
                                 __bf16* __restrict__ dst, int K, int N) {
  int i = blockIdx.x * blockDim.x + threadIdx.x;
  if (i >= K * N) return;
  int n = i / K, k = i - n * K;
  dst[(size_t)n * K + k] = (__bf16)src[(size_t)k * N + n];
}

// ---------------------------------------------------------------------------
// Projection GEMM: Y(16384 x 256) = X(16384 x K) @ W(K x 256) + bias
// WT is bf16 column-major (256 x K).  grid.x = 1024 M-tiles, 256 thr (8 waves)
// A-tile (16 rows x K, contiguous in X) staged once per block to LDS via
// async global->LDS; each wave computes N columns [32w, 32w+32).
// ---------------------------------------------------------------------------
__global__ void k_proj_gemm(const float* __restrict__ X,
                            const __bf16* __restrict__ WT,
                            const float* __restrict__ bias,
                            float* __restrict__ Y, int K) {
  __shared__ __align__(16) float As[16 * KMAX];

  // Stage A tile: one contiguous chunk of 16*K floats.
  const float* src = X + (size_t)blockIdx.x * 16 * K;
  for (int i = threadIdx.x * 4; i < 16 * K; i += 256 * 4)
    async_copy_b128(src + i, &As[i]);
  wait_asynccnt0();
  __syncthreads();

  const int lane  = threadIdx.x & 31;
  const int wave  = threadIdx.x >> 5;
  const int khalf = (lane >> 4) << 4;            // 0 / 16 : lane-half K split
  const int col0  = wave * 32 + (lane & 15);
  const int col1  = col0 + 16;

  const float* arow = As + (lane & 15) * K + khalf;
  v8f acc0 = {}; v8f acc1 = {};

  for (int kk = 0; kk < K; kk += 32) {
    v16bf a  = frag_from_f32(arow + kk);
    v16bf b0 = *(const v16bf*)(WT + (size_t)col0 * K + kk + khalf);
    v16bf b1 = *(const v16bf*)(WT + (size_t)col1 * K + kk + khalf);
    acc0 = wmma_bf16(a, b0, acc0);
    acc1 = wmma_bf16(a, b1, acc1);
  }
  const float bc0 = bias[col0], bc1 = bias[col1];
#pragma unroll
  for (int r = 0; r < 8; ++r) {
    int orow = blockIdx.x * 16 + r + ((lane >> 4) << 3);
    Y[(size_t)orow * Hdim + col0] = acc0[r] + bc0;
    Y[(size_t)orow * Hdim + col1] = acc1[r] + bc1;
  }
}

// ---------------------------------------------------------------------------
// Attention scores: 3 pair attentions + 3 unimodal, raw (pre-softmax).
// scores layout: [ch][b*1024+t], ch = 0..2 pair, 3..5 uni.
// ---------------------------------------------------------------------------
__global__ void k_att_scores(const float* __restrict__ tr,
                             const float* __restrict__ fr,
                             const float* __restrict__ ar,
                             const float* __restrict__ attW,
                             const float* __restrict__ attb,
                             const float* __restrict__ utW, const float* __restrict__ utb,
                             const float* __restrict__ ufW, const float* __restrict__ ufb,
                             const float* __restrict__ uaW, const float* __restrict__ uab,
                             float* __restrict__ scores) {
  int idx = blockIdx.x * blockDim.x + threadIdx.x;   // b*1024 + t
  if (idx >= Bsz * Tlen) return;
  const float* tp = tr + (size_t)idx * Hdim;
  const float* fp = fr + (size_t)idx * Hdim;
  const float* ap = ar + (size_t)idx * Hdim;
  float s1 = attb[0], s2 = attb[0], s3 = attb[0];
  float su_t = utb[0], su_f = ufb[0], su_a = uab[0];
  for (int h = 0; h < Hdim; ++h) {
    float th = tanhf(tp[h]), fh = tanhf(fp[h]), ah = tanhf(ap[h]);
    float w0 = attW[h], w1 = attW[Hdim + h];
    s1 += th * w0 + fh * w1;     // tanh(cat(tr,fr)) @ attW
    s2 += fh * w0 + ah * w1;     // tanh(cat(fr,ar)) @ attW
    s3 += th * w0 + ah * w1;     // tanh(cat(tr,ar)) @ attW
    su_t += th * utW[h];  su_f += fh * ufW[h];  su_a += ah * uaW[h];
  }
  scores[0 * Bsz * Tlen + idx] = s1;
  scores[1 * Bsz * Tlen + idx] = s2;
  scores[2 * Bsz * Tlen + idx] = s3;
  scores[3 * Bsz * Tlen + idx] = su_t;
  scores[4 * Bsz * Tlen + idx] = su_f;
  scores[5 * Bsz * Tlen + idx] = su_a;
}

// ---------------------------------------------------------------------------
// Softmax over TIME per (channel, batch).  grid = 6*16 blocks, 256 threads.
// ch<3 -> attn[b,t,ch] ; ch>=3 -> enc_input[b,t,ch-3]
// ---------------------------------------------------------------------------
__global__ void k_softmax_T(const float* __restrict__ scores,
                            float* __restrict__ attn,
                            float* __restrict__ enc_input) {
  const int ch = blockIdx.x >> 4, b = blockIdx.x & 15;
  const float* s = scores + (size_t)ch * Bsz * Tlen + (size_t)b * Tlen;
  __shared__ float red[256];
  const int tid = threadIdx.x;

  float m = -3.4e38f;
  for (int t = tid; t < Tlen; t += 256) m = fmaxf(m, s[t]);
  red[tid] = m; __syncthreads();
  for (int k = 128; k > 0; k >>= 1) {
    if (tid < k) red[tid] = fmaxf(red[tid], red[tid + k]);
    __syncthreads();
  }
  const float mx = red[0]; __syncthreads();

  float sum = 0.f;
  for (int t = tid; t < Tlen; t += 256) sum += __expf(s[t] - mx);
  red[tid] = sum; __syncthreads();
  for (int k = 128; k > 0; k >>= 1) {
    if (tid < k) red[tid] += red[tid + k];
    __syncthreads();
  }
  const float inv = 1.0f / red[0];

  for (int t = tid; t < Tlen; t += 256) {
    float v = __expf(s[t] - mx) * inv;
    if (ch < 3) attn[((size_t)b * Tlen + t) * 3 + ch]          = v;
    else        enc_input[((size_t)b * Tlen + t) * 3 + (ch-3)] = v;
  }
}

// ---------------------------------------------------------------------------
// Encoder cLSTM: 3 independent LSTMs (blockIdx.x = l), persistent workgroup.
// Per step: gates = x@Wih + b  (scalar, input dim 3), then h@Whh via WMMA,
// gate nonlinearity (c in registers), h -> LDS bf16, enc_out dot.
// ---------------------------------------------------------------------------
__global__ void k_enc_lstm(const float* __restrict__ enc_input,   // (B,T,3)
                           const float* __restrict__ Wih,         // (3,3,1024)
                           const float* __restrict__ bias,        // (3,1024)
                           const __bf16* __restrict__ WhhT,       // (3,1024,256)
                           const float* __restrict__ outW,        // (3,256)
                           const float* __restrict__ outb,        // (3)
                           float* __restrict__ enc_out) {         // (B,T,3)
  const int l    = blockIdx.x;
  const int tid  = threadIdx.x;
  const int lane = tid & 31;
  const int wave = tid >> 5;

  __shared__ __bf16 gS[Bsz * G4H];   // 32 KB gate pre-activations
  __shared__ __bf16 hB[Bsz * Hdim];  //  8 KB hidden state (bf16)
  __shared__ float  red[256];

  const float*  Wl  = Wih  + (size_t)l * 3 * G4H;
  const float*  bl  = bias + (size_t)l * G4H;
  const __bf16* Whl = WhhT + (size_t)l * G4H * Hdim;
  const float*  ow  = outW + (size_t)l * Hdim;
  const float   ob  = outb[l];

  // c state in registers: thread owns b = tid&15, h = (tid>>4)*16 + j
  const int cb = tid & 15, ch0 = (tid >> 4) << 4;
  float c[16];
#pragma unroll
  for (int j = 0; j < 16; ++j) c[j] = 0.f;
  for (int i = tid; i < Bsz * Hdim; i += 256) hB[i] = (__bf16)0.f;
  __syncthreads();

  for (int t = 0; t < Tlen; ++t) {
    // 1) gates pre-activation from 3-dim input
    for (int i = tid; i < Bsz * G4H; i += 256) {
      int b = i >> 10, j = i & (G4H - 1);
      const float* xp = enc_input + ((size_t)b * Tlen + t) * 3;
      float v = bl[j] + xp[0] * Wl[j] + xp[1] * Wl[G4H + j] + xp[2] * Wl[2 * G4H + j];
      gS[i] = (__bf16)v;
    }
    __syncthreads();

    // 2) gates += h @ Whh   (16 x 1024 = 64 N-tiles, 8 per wave, K=256)
    {
      const int khalf = (lane >> 4) << 4;
      const int rbase = (lane >> 4) << 3;
#pragma unroll
      for (int nt = 0; nt < 8; ++nt) {
        const int ntile = wave * 8 + nt;
        const int col   = ntile * 16 + (lane & 15);
        v8f acc;
#pragma unroll
        for (int r = 0; r < 8; ++r) acc[r] = (float)gS[(size_t)(rbase + r) * G4H + col];
        const __bf16* wcol = Whl + (size_t)col * Hdim;
#pragma unroll
        for (int kk = 0; kk < Hdim; kk += 32) {
          v16bf a = *(const v16bf*)(hB + (size_t)(lane & 15) * Hdim + kk + khalf);
          v16bf b = *(const v16bf*)(wcol + kk + khalf);
          acc = wmma_bf16(a, b, acc);
        }
#pragma unroll
        for (int r = 0; r < 8; ++r) gS[(size_t)(rbase + r) * G4H + col] = (__bf16)acc[r];
      }
    }
    __syncthreads();

    // 3) gate nonlinearity, update c (regs) and h (LDS bf16)
#pragma unroll
    for (int j = 0; j < 16; ++j) {
      int h = ch0 + j;
      float gi = (float)gS[(size_t)cb * G4H + h];
      float gf = (float)gS[(size_t)cb * G4H + Hdim + h];
      float gg = (float)gS[(size_t)cb * G4H + 2 * Hdim + h];
      float go = (float)gS[(size_t)cb * G4H + 3 * Hdim + h];
      c[j] = sigm(gf) * c[j] + sigm(gi) * tanhf(gg);
      hB[(size_t)cb * Hdim + h] = (__bf16)(sigm(go) * tanhf(c[j]));
    }
    __syncthreads();

    // 4) enc_out[b,t,l] = dot(h[b], outW) + ob : 16-way partial + reduce
    {
      const int b = tid >> 4, chunk = (tid & 15) << 4;
      float p = 0.f;
#pragma unroll
      for (int e = 0; e < 16; ++e)
        p += (float)hB[(size_t)b * Hdim + chunk + e] * ow[chunk + e];
      red[tid] = p; __syncthreads();
      if (tid < Bsz) {
        float s = ob;
#pragma unroll
        for (int e = 0; e < 16; ++e) s += red[tid * 16 + e];
        enc_out[((size_t)tid * Tlen + t) * 3 + l] = s;
      }
      __syncthreads();
    }
  }
}

// ---------------------------------------------------------------------------
// context[b,t,k] = sum_i attn[b,t,i] * enc_out[b,t-i,k]  (causal, zero pad)
// ---------------------------------------------------------------------------
__global__ void k_context(const float* __restrict__ attn,
                          const float* __restrict__ enc_out,
                          float* __restrict__ context) {
  int i = blockIdx.x * blockDim.x + threadIdx.x;
  if (i >= Bsz * Tlen * 3) return;
  int k = i % 3, bt = i / 3, t = bt % Tlen, b = bt / Tlen;
  float acc = 0.f;
#pragma unroll
  for (int s = 0; s < 3; ++s) {
    float e = (t - s >= 0) ? enc_out[(((size_t)b * Tlen) + (t - s)) * 3 + k] : 0.f;
    acc += attn[(size_t)bt * 3 + s] * e;
  }
  context[i] = acc;
}

// ---------------------------------------------------------------------------
// Decoder LSTM (teacher forcing) + output head.  Single persistent block.
// ---------------------------------------------------------------------------
__global__ void k_dec_lstm(const float* __restrict__ target,   // (B,T,1)
                           const float* __restrict__ context,  // (B,T,3)
                           const float* __restrict__ Wih,      // (4,1024)
                           const float* __restrict__ bias,     // (1024)
                           const __bf16* __restrict__ WhhT,    // (1024,256)
                           const float* __restrict__ h0,
                           const float* __restrict__ c0,
                           const float* __restrict__ o1W,      // (256,4)
                           const float* __restrict__ o1b,      // (4)
                           const float* __restrict__ o2W,      // (4,1)
                           const float* __restrict__ o2b,      // (1)
                           float* __restrict__ out) {          // (B,T,1)
  const int tid  = threadIdx.x;
  const int lane = tid & 31;
  const int wave = tid >> 5;

  __shared__ __bf16 gS[Bsz * G4H];
  __shared__ __bf16 hB[Bsz * Hdim];
  __shared__ float  red4[256 * 4];

  const int cb = tid & 15, ch0 = (tid >> 4) << 4;
  float c[16];
#pragma unroll
  for (int j = 0; j < 16; ++j) c[j] = c0[ch0 + j];
  for (int i = tid; i < Bsz * Hdim; i += 256) hB[i] = (__bf16)h0[i & (Hdim - 1)];
  __syncthreads();

  for (int t = 0; t < Tlen; ++t) {
    for (int i = tid; i < Bsz * G4H; i += 256) {
      int b = i >> 10, j = i & (G4H - 1);
      float x0 = (t > 0) ? target[(size_t)b * Tlen + (t - 1)] : 0.f;
      const float* cp = context + ((size_t)b * Tlen + t) * 3;
      float v = bias[j] + x0 * Wih[j] + cp[0] * Wih[G4H + j]
              + cp[1] * Wih[2 * G4H + j] + cp[2] * Wih[3 * G4H + j];
      gS[i] = (__bf16)v;
    }
    __syncthreads();

    {
      const int khalf = (lane >> 4) << 4;
      const int rbase = (lane >> 4) << 3;
#pragma unroll
      for (int nt = 0; nt < 8; ++nt) {
        const int ntile = wave * 8 + nt;
        const int col   = ntile * 16 + (lane & 15);
        v8f acc;
#pragma unroll
        for (int r = 0; r < 8; ++r) acc[r] = (float)gS[(size_t)(rbase + r) * G4H + col];
        const __bf16* wcol = WhhT + (size_t)col * Hdim;
#pragma unroll
        for (int kk = 0; kk < Hdim; kk += 32) {
          v16bf a = *(const v16bf*)(hB + (size_t)(lane & 15) * Hdim + kk + khalf);
          v16bf b = *(const v16bf*)(wcol + kk + khalf);
          acc = wmma_bf16(a, b, acc);
        }
#pragma unroll
        for (int r = 0; r < 8; ++r) gS[(size_t)(rbase + r) * G4H + col] = (__bf16)acc[r];
      }
    }
    __syncthreads();

#pragma unroll
    for (int j = 0; j < 16; ++j) {
      int h = ch0 + j;
      float gi = (float)gS[(size_t)cb * G4H + h];
      float gf = (float)gS[(size_t)cb * G4H + Hdim + h];
      float gg = (float)gS[(size_t)cb * G4H + 2 * Hdim + h];
      float go = (float)gS[(size_t)cb * G4H + 3 * Hdim + h];
      c[j] = sigm(gf) * c[j] + sigm(gi) * tanhf(gg);
      hB[(size_t)cb * Hdim + h] = (__bf16)(sigm(go) * tanhf(c[j]));
    }
    __syncthreads();

    // output head: relu(h @ o1W + o1b) @ o2W + o2b
    {
      const int b = tid >> 4, chunk = (tid & 15) << 4;
      float p0 = 0.f, p1 = 0.f, p2 = 0.f, p3 = 0.f;
#pragma unroll
      for (int e = 0; e < 16; ++e) {
        float hv = (float)hB[(size_t)b * Hdim + chunk + e];
        const float* wr = o1W + (size_t)(chunk + e) * 4;
        p0 += hv * wr[0]; p1 += hv * wr[1]; p2 += hv * wr[2]; p3 += hv * wr[3];
      }
      red4[tid * 4 + 0] = p0; red4[tid * 4 + 1] = p1;
      red4[tid * 4 + 2] = p2; red4[tid * 4 + 3] = p3;
      __syncthreads();
      if (tid < Bsz) {
        float q[4] = {o1b[0], o1b[1], o1b[2], o1b[3]};
#pragma unroll
        for (int e = 0; e < 16; ++e) {
#pragma unroll
          for (int qq = 0; qq < 4; ++qq) q[qq] += red4[(tid * 16 + e) * 4 + qq];
        }
        float y = o2b[0];
#pragma unroll
        for (int qq = 0; qq < 4; ++qq) y += fmaxf(q[qq], 0.f) * o2W[qq];
        out[(size_t)tid * Tlen + t] = y;
      }
      __syncthreads();
    }
  }
}

// ---------------------------------------------------------------------------
extern "C" void kernel_launch(void* const* d_in, const int* in_sizes, int n_in,
                              void* d_out, int out_size, void* d_ws, size_t ws_size,
                              hipStream_t stream) {
  (void)in_sizes; (void)n_in; (void)out_size; (void)ws_size;

  const float* face   = (const float*)d_in[1];
  const float* audio  = (const float*)d_in[2];
  const float* text   = (const float*)d_in[3];
  const float* target = (const float*)d_in[4];
  const float* text_W = (const float*)d_in[5];   const float* text_b = (const float*)d_in[6];
  const float* face_W = (const float*)d_in[7];   const float* face_b = (const float*)d_in[8];
  const float* audio_W= (const float*)d_in[9];   const float* audio_b= (const float*)d_in[10];
  const float* att_W  = (const float*)d_in[11];  const float* att_b  = (const float*)d_in[12];
  const float* ut_W   = (const float*)d_in[13];  const float* ut_b   = (const float*)d_in[14];
  const float* uf_W   = (const float*)d_in[15];  const float* uf_b   = (const float*)d_in[16];
  const float* ua_W   = (const float*)d_in[17];  const float* ua_b   = (const float*)d_in[18];
  const float* enc_Wih= (const float*)d_in[19];
  const float* enc_Whh= (const float*)d_in[20];
  const float* enc_b  = (const float*)d_in[21];
  const float* enc_oW = (const float*)d_in[22];  const float* enc_ob = (const float*)d_in[23];
  const float* dec_Wih= (const float*)d_in[24];
  const float* dec_Whh= (const float*)d_in[25];
  const float* dec_b  = (const float*)d_in[26];
  const float* dec_h0 = (const float*)d_in[27];  const float* dec_c0 = (const float*)d_in[28];
  const float* o1W    = (const float*)d_in[29];  const float* o1b    = (const float*)d_in[30];
  const float* o2W    = (const float*)d_in[31];  const float* o2b    = (const float*)d_in[32];

  char* ws = (char*)d_ws;
  float*  tr   = (float*)(ws + OFF_TR);
  float*  fr   = (float*)(ws + OFF_FR);
  float*  ar   = (float*)(ws + OFF_AR);
  float*  sc   = (float*)(ws + OFF_SC);
  float*  ein  = (float*)(ws + OFF_EIN);
  float*  attn = (float*)(ws + OFF_ATT);
  float*  eo   = (float*)(ws + OFF_EO);
  float*  ctx  = (float*)(ws + OFF_CTX);
  __bf16* WtT  = (__bf16*)(ws + OFF_WTT);
  __bf16* WfT  = (__bf16*)(ws + OFF_WFT);
  __bf16* WaT  = (__bf16*)(ws + OFF_WAT);
  __bf16* WhE  = (__bf16*)(ws + OFF_WHE);
  __bf16* WhD  = (__bf16*)(ws + OFF_WHD);

  // 1) bf16 transposed weights (B-fragments become contiguous 32B loads)
  k_transpose_bf16<<<(768*256+255)/256, 256, 0, stream>>>(text_W,  WtT, 768, 256);
  k_transpose_bf16<<<(512*256+255)/256, 256, 0, stream>>>(face_W,  WfT, 512, 256);
  k_transpose_bf16<<<(256*256+255)/256, 256, 0, stream>>>(audio_W, WaT, 256, 256);
  for (int l = 0; l < 3; ++l)
    k_transpose_bf16<<<(256*1024+255)/256, 256, 0, stream>>>(
        enc_Whh + (size_t)l * 256 * 1024, WhE + (size_t)l * 1024 * 256, 256, 1024);
  k_transpose_bf16<<<(256*1024+255)/256, 256, 0, stream>>>(dec_Whh, WhD, 256, 1024);

  // 2) modality projections (WMMA bf16, async-LDS staged A tiles)
  k_proj_gemm<<<1024, 256, 0, stream>>>(text,  WtT, text_b,  tr, 768);
  k_proj_gemm<<<1024, 256, 0, stream>>>(face,  WfT, face_b,  fr, 512);
  k_proj_gemm<<<1024, 256, 0, stream>>>(audio, WaT, audio_b, ar, 256);

  // 3) attention scores + softmax over time
  k_att_scores<<<(Bsz*Tlen+255)/256, 256, 0, stream>>>(
      tr, fr, ar, att_W, att_b, ut_W, ut_b, uf_W, uf_b, ua_W, ua_b, sc);
  k_softmax_T<<<6*Bsz, 256, 0, stream>>>(sc, attn, ein);

  // 4) encoder cLSTM: 3 persistent workgroups, WMMA recurrent matmul
  k_enc_lstm<<<3, 256, 0, stream>>>(ein, enc_Wih, enc_b, WhE, enc_oW, enc_ob, eo);

  // 5) causal attention-weighted context
  k_context<<<(Bsz*Tlen*3+255)/256, 256, 0, stream>>>(attn, eo, ctx);

  // 6) decoder LSTM + output head
  k_dec_lstm<<<1, 256, 0, stream>>>(target, ctx, dec_Wih, dec_b, WhD,
                                    dec_h0, dec_c0, o1W, o1b, o2W, o2b,
                                    (float*)d_out);
}
